// attention_b_32547262169743
// MI455X (gfx1250) — matrix operand order repaired and checked by reading, not compiled, not run on hardware
//
#include <hip/hip_runtime.h>
#include <hip/hip_bf16.h>
#include <math.h>

typedef __bf16 bf16_t;
typedef bf16_t v16bf __attribute__((ext_vector_type(16)));
typedef bf16_t v8bf  __attribute__((ext_vector_type(8)));
typedef float  v8f   __attribute__((ext_vector_type(8)));

#define N_B   64
#define C_CH  192
#define T_DIM 128
#define V_DIM 25
#define P_DIM (T_DIM * V_DIM)       // 3200
#define O3    (3 * C_CH)            // 576
#define MID   64
#define ATT_OFF ((size_t)N_B * C_CH * P_DIM)   // 39321600

__device__ inline v16bf cat8(v8bf lo, v8bf hi) {
  return __builtin_shufflevector(lo, hi, 0,1,2,3,4,5,6,7,8,9,10,11,12,13,14,15);
}

// ---- prep: bf16 weights + folded BN affine -------------------------------
__global__ __launch_bounds__(256) void prep_kernel(
    const float* __restrict__ Wq, const float* __restrict__ Wf,
    const float* __restrict__ gamma, const float* __restrict__ beta,
    const float* __restrict__ mean,  const float* __restrict__ var,
    bf16_t* __restrict__ Wq_bf, bf16_t* __restrict__ Wf_bf,
    float* __restrict__ inv, float* __restrict__ shf)
{
  int i = blockIdx.x * 256 + threadIdx.x;
  const int n1 = O3 * C_CH;            // 110592
  const int n2 = n1 + C_CH * C_CH;     // 147456
  if (i < n1) {
    Wq_bf[i] = (bf16_t)Wq[i];
  } else if (i < n2) {
    int j = i - n1;
    Wf_bf[j] = (bf16_t)Wf[j];
  } else if (i < n2 + C_CH) {
    int c = i - n2;
    float iv = gamma[c] / sqrtf(var[c] + 1e-5f);
    inv[c] = iv;
    shf[c] = beta[c] - mean[c] * iv;
  }
}

// ---- transpose x: (n, c, p) f32 -> (n, p, c) bf16 ------------------------
__global__ __launch_bounds__(256) void transpose_x_kernel(
    const float* __restrict__ x, bf16_t* __restrict__ xt)
{
  __shared__ float tile[32][33];
  int b  = blockIdx.x;                 // 64 * 6 * 100 = 38400
  int n  = b / 600;
  int rr = b % 600;
  int c0 = (rr / 100) * 32;
  int p0 = (rr % 100) * 32;
  int tx = threadIdx.x & 31, ty = threadIdx.x >> 5;   // 32 x 8
  const float* xb = x + (size_t)n * C_CH * P_DIM;
#pragma unroll
  for (int i = 0; i < 4; i++)
    tile[ty + 8*i][tx] = xb[(size_t)(c0 + ty + 8*i) * P_DIM + p0 + tx];
  __syncthreads();
  bf16_t* xo = xt + (size_t)n * P_DIM * C_CH;
#pragma unroll
  for (int i = 0; i < 4; i++)
    xo[(size_t)(p0 + ty + 8*i) * C_CH + c0 + tx] = (bf16_t)tile[tx][ty + 8*i];
}

// ---- QKV GEMM: qkv[n,o,p] = Wqkv[o,:]·x[n,:,p] + b, bf16 out (WMMA) ------
__global__ __launch_bounds__(256) void qkv_gemm_kernel(
    const bf16_t* __restrict__ Wq, const bf16_t* __restrict__ xt,
    const float* __restrict__ bq, bf16_t* __restrict__ qkv)
{
  int wg   = blockIdx.x * 8 + (threadIdx.x >> 5);   // 115200 waves
  int lane = threadIdx.x & 31;
  int n  = wg / 1800;
  int r  = wg % 1800;
  int o0 = (r / 50) * 16;
  int p0 = (r % 50) * 64;
  int lo = lane & 15, kh = lane >> 4;

  const bf16_t* xn = xt + (size_t)n * P_DIM * C_CH;
  v8f acc[4];
#pragma unroll
  for (int j = 0; j < 4; j++) acc[j] = (v8f){0.f,0.f,0.f,0.f,0.f,0.f,0.f,0.f};

#pragma unroll
  for (int ks = 0; ks < C_CH; ks += 32) {
    const bf16_t* ap = Wq + (size_t)(o0 + lo) * C_CH + ks + kh * 8;
    v16bf a = cat8(*(const v8bf*)ap, *(const v8bf*)(ap + 16));
#pragma unroll
    for (int j = 0; j < 4; j++) {
      const bf16_t* bp = xn + (size_t)(p0 + j*16 + lo) * C_CH + ks + kh * 8;
      v16bf b = cat8(*(const v8bf*)bp, *(const v8bf*)(bp + 16));
      acc[j] = __builtin_amdgcn_wmma_f32_16x16x32_bf16(
          false, a, false, b, (short)0, acc[j], false, false);
    }
  }

  size_t base = (size_t)n * O3 * P_DIM;
#pragma unroll
  for (int j = 0; j < 4; j++) {
    int p = p0 + j*16 + lo;
#pragma unroll
    for (int rr2 = 0; rr2 < 8; rr2++) {
      int o = o0 + rr2 + kh * 8;
      float v = acc[j][rr2] + bq[o];
      qkv[base + (size_t)o * P_DIM + p] = (bf16_t)v;
    }
  }
}

// ---- attention: att[u,v] = tanh( sum_{c',t} q·k / 8192 ) -----------------
__global__ __launch_bounds__(256) void att_kernel(
    const bf16_t* __restrict__ qkv, float* __restrict__ att_out)
{
  __shared__ float qs[64 * 25];
  __shared__ float ksh[64 * 25];
  int ns = blockIdx.x;                  // 192 = N*S
  int n = ns / 3, s = ns % 3;
  const bf16_t* qb = qkv + (size_t)n * O3 * P_DIM + (size_t)(s * MID) * P_DIM;
  const bf16_t* kb = qkv + (size_t)n * O3 * P_DIM + (size_t)((3 + s) * MID) * P_DIM;
  int tid = threadIdx.x;
  float a0 = 0.f, a1 = 0.f, a2 = 0.f;
  int p0 = tid, p1 = tid + 256, p2 = tid + 512;
  int u0 = p0 / 25, v0 = p0 % 25;
  int u1 = p1 / 25, v1 = p1 % 25;
  int u2 = (p2 < 625) ? p2 / 25 : 0, v2 = (p2 < 625) ? p2 % 25 : 0;

  for (int kk = 0; kk < MID * T_DIM; kk += 64) {
    __syncthreads();
    for (int i = tid; i < 1600; i += 256) {
      qs[i]  = (float)qb[(size_t)kk * 25 + i];
      ksh[i] = (float)kb[(size_t)kk * 25 + i];
    }
    __syncthreads();
#pragma unroll 4
    for (int j = 0; j < 64; j++) {
      const float* qr = &qs[j * 25];
      const float* kr = &ksh[j * 25];
      a0 += qr[u0] * kr[v0];
      a1 += qr[u1] * kr[v1];
      a2 += qr[u2] * kr[v2];
    }
  }
  const float sc = 1.0f / 8192.0f;
  float* ob = att_out + (size_t)ns * 625;
  ob[p0] = tanhf(a0 * sc);
  ob[p1] = tanhf(a1 * sc);
  if (p2 < 625) ob[p2] = tanhf(a2 * sc);
}

// ---- aggregation: y_t[n,p,c] (bf16) = sum_v att[u,v]*vmat[c',t,v] --------
__global__ __launch_bounds__(256) void agg_kernel(
    const bf16_t* __restrict__ qkv, const float* __restrict__ att,
    bf16_t* __restrict__ yt)
{
  __shared__ float a_s[625];
  int ns = blockIdx.x;
  int n = ns / 3, s = ns % 3;
  int tid = threadIdx.x;
  for (int i = tid; i < 625; i += 256) a_s[i] = att[(size_t)ns * 625 + i];
  __syncthreads();
  int cp = tid & 63;        // c'
  int tg = tid >> 6;        // 0..3
  const bf16_t* vb = qkv + (size_t)n * O3 * P_DIM
                         + (size_t)((6 + s) * MID + cp) * P_DIM;
  bf16_t* yb = yt + (size_t)n * P_DIM * C_CH + (s * MID + cp);
  for (int t = tg; t < T_DIM; t += 4) {
    float vr[25];
#pragma unroll
    for (int vv = 0; vv < 25; vv++) vr[vv] = (float)vb[t * 25 + vv];
#pragma unroll
    for (int u = 0; u < 25; u++) {
      float acc = 0.f;
#pragma unroll
      for (int vv = 0; vv < 25; vv++) acc += a_s[u * 25 + vv] * vr[vv];
      yb[(size_t)(t * 25 + u) * C_CH] = (bf16_t)acc;
    }
  }
}

// ---- FF GEMM + bias + BN + residual + LeakyReLU (WMMA) -------------------
__global__ __launch_bounds__(256) void ff_gemm_kernel(
    const bf16_t* __restrict__ Wf, const bf16_t* __restrict__ yt,
    const float* __restrict__ bf, const float* __restrict__ inv,
    const float* __restrict__ shf, const float* __restrict__ x,
    float* __restrict__ out)
{
  int wg   = blockIdx.x * 8 + (threadIdx.x >> 5);   // 38400 waves
  int lane = threadIdx.x & 31;
  int n  = wg / 600;
  int r  = wg % 600;
  int o0 = (r / 50) * 16;
  int p0 = (r % 50) * 64;
  int lo = lane & 15, kh = lane >> 4;

  const bf16_t* yn = yt + (size_t)n * P_DIM * C_CH;
  v8f acc[4];
#pragma unroll
  for (int j = 0; j < 4; j++) acc[j] = (v8f){0.f,0.f,0.f,0.f,0.f,0.f,0.f,0.f};

#pragma unroll
  for (int ks = 0; ks < C_CH; ks += 32) {
    const bf16_t* ap = Wf + (size_t)(o0 + lo) * C_CH + ks + kh * 8;
    v16bf a = cat8(*(const v8bf*)ap, *(const v8bf*)(ap + 16));
#pragma unroll
    for (int j = 0; j < 4; j++) {
      const bf16_t* bp = yn + (size_t)(p0 + j*16 + lo) * C_CH + ks + kh * 8;
      v16bf b = cat8(*(const v8bf*)bp, *(const v8bf*)(bp + 16));
      acc[j] = __builtin_amdgcn_wmma_f32_16x16x32_bf16(
          false, a, false, b, (short)0, acc[j], false, false);
    }
  }

  size_t xb = (size_t)n * C_CH * P_DIM;
#pragma unroll
  for (int j = 0; j < 4; j++) {
    int p = p0 + j*16 + lo;
#pragma unroll
    for (int rr2 = 0; rr2 < 8; rr2++) {
      int o = o0 + rr2 + kh * 8;
      float z = acc[j][rr2] + bf[o];
      z = z * inv[o] + shf[o];
      float res = x[xb + (size_t)o * P_DIM + p] + z;
      out[xb + (size_t)o * P_DIM + p] = res > 0.f ? res : 0.1f * res;
    }
  }
}

extern "C" void kernel_launch(void* const* d_in, const int* in_sizes, int n_in,
                              void* d_out, int out_size, void* d_ws, size_t ws_size,
                              hipStream_t stream) {
  const float* x     = (const float*)d_in[0];
  const float* Wq    = (const float*)d_in[1];
  const float* bq    = (const float*)d_in[2];
  const float* Wf    = (const float*)d_in[3];
  const float* bf    = (const float*)d_in[4];
  const float* gamma = (const float*)d_in[5];
  const float* beta  = (const float*)d_in[6];
  const float* mean  = (const float*)d_in[7];
  const float* var   = (const float*)d_in[8];
  float* out = (float*)d_out;

  char* ws = (char*)d_ws;
  size_t off = 0;
  auto wsalloc = [&](size_t bytes) -> void* {
    void* p = ws + off;
    off = (off + bytes + 255) & ~(size_t)255;
    return p;
  };
  bf16_t* xt    = (bf16_t*)wsalloc((size_t)N_B * P_DIM * C_CH * 2);  // 78.6 MB
  bf16_t* qkv   = (bf16_t*)wsalloc((size_t)N_B * O3 * P_DIM * 2);    // 236 MB
  bf16_t* yt    = (bf16_t*)wsalloc((size_t)N_B * P_DIM * C_CH * 2);  // 78.6 MB
  bf16_t* Wq_bf = (bf16_t*)wsalloc((size_t)O3 * C_CH * 2);
  bf16_t* Wf_bf = (bf16_t*)wsalloc((size_t)C_CH * C_CH * 2);
  float*  inv   = (float*)wsalloc(C_CH * 4);
  float*  shf   = (float*)wsalloc(C_CH * 4);

  float* att_out = out + ATT_OFF;

  prep_kernel<<<577, 256, 0, stream>>>(Wq, Wf, gamma, beta, mean, var,
                                       Wq_bf, Wf_bf, inv, shf);
  transpose_x_kernel<<<38400, 256, 0, stream>>>(x, xt);
  qkv_gemm_kernel<<<14400, 256, 0, stream>>>(Wq_bf, xt, bq, qkv);
  att_kernel<<<192, 256, 0, stream>>>(qkv, att_out);
  agg_kernel<<<192, 256, 0, stream>>>(qkv, att_out, yt);
  ff_gemm_kernel<<<4800, 256, 0, stream>>>(Wf_bf, yt, bf, inv, shf, x, out);
}
